// Nystromformer_56100862820684
// MI455X (gfx1250) — matrix operand order repaired and checked by reading, compiled, not run on hardware
//
#include <hip/hip_runtime.h>
#include <math.h>

// ---------------------------------------------------------------------------
// Nystromformer attention for MI455X (gfx1250), wave32, WMMA.
//   b=4 h=16 s=4096 d=64, m=64 landmarks, 6 pinv iterations.
// Big s-dim GEMMs: v_wmma_f32_16x16x32_f16 (f16 operands, f32 accum).
// Pinv Newton-Schulz GEMMs: v_wmma_f32_16x16x4_f32 (exact fp32).
// All LDS operand layouts arranged so WMMA fragment loads are contiguous
// (merge into ds_load_b128); bulk global traffic uses float4 (b128).
// ---------------------------------------------------------------------------

typedef __attribute__((ext_vector_type(16))) _Float16 v16h;
typedef __attribute__((ext_vector_type(8)))  float    v8f;
typedef __attribute__((ext_vector_type(2)))  float    v2f;

#define NBH    64      // b*h
#define SEQ    4096
#define DIM    64
#define SCALE  0.125f  // 1/sqrt(64)
#define BVSPLIT 8      // workgroups per head for the BV streaming pass
#define TP     65      // padded pitch (halves) for transposed LDS operands

// ---------------- WMMA wrappers -------------------------------------------
static __device__ __forceinline__ v8f wmma_f16(v16h a, v16h b, v8f c) {
  return __builtin_amdgcn_wmma_f32_16x16x32_f16(false, a, false, b, (short)0, c,
                                                false, false);
}
static __device__ __forceinline__ v8f wmma_f32k4(v2f a, v2f b, v8f c) {
  return __builtin_amdgcn_wmma_f32_16x16x4_f32(false, a, false, b, (short)0, c,
                                               false, false);
}

// A fragment, 16x32 f16, operand stored row-major [M][64] in LDS.
// lane<16: halves 0..15 -> K 0..15, hi lanes K 16..31.
static __device__ __forceinline__ v16h ldA(const _Float16* Amk, int ti, int kb,
                                           int mrow, int hi) {
  v16h a;
  const _Float16* r = Amk + (ti * 16 + mrow) * 64 + kb;
#pragma unroll
  for (int i = 0; i < 16; ++i) a[i] = r[hi * 16 + i];
  return a;
}
// B fragment, 32x16 f16, operand stored [N][K] row-major with pitch `ld`
// (i.e. B[k][n] = Bnk[n*ld + k]); lane<16 holds K 0..15, hi lanes K 16..31.
static __device__ __forceinline__ v16h ldB_nk(const _Float16* Bnk, int tj, int kb,
                                              int ncol, int hi, int ld) {
  v16h b;
  const _Float16* r = Bnk + (tj * 16 + ncol) * ld + kb + hi * 16;
#pragma unroll
  for (int i = 0; i < 16; ++i) b[i] = r[i];
  return b;
}

// 64x64x64 fp32 GEMM in LDS via v_wmma_f32_16x16x4_f32.
// mode 0: dst = A@B   mode 1: dst = beta*I - A@B   mode 2: dst = 0.25*(A@B)
static __device__ __forceinline__ void mm64_f32(float* __restrict__ dst,
                                                const float* __restrict__ A,
                                                const float* __restrict__ B,
                                                int mode, float beta) {
  const int t = threadIdx.x, lane = t & 31, wave = t >> 5;
  const int mrow = lane & 15, hi = lane >> 4;
#pragma unroll
  for (int tt = 0; tt < 2; ++tt) {
    const int idx = wave + tt * 8, ti = idx >> 2, tj = idx & 3;
    v8f acc = {};
#pragma unroll
    for (int kk = 0; kk < 16; ++kk) {
      const int k0 = kk * 4 + hi * 2;  // f32 A: lane<16 K{k0,k0+1}, hi K{k0+2,k0+3}
      v2f av, bv;
      av.x = A[(ti * 16 + mrow) * 64 + k0];
      av.y = A[(ti * 16 + mrow) * 64 + k0 + 1];
      bv.x = B[k0 * 64 + tj * 16 + mrow];
      bv.y = B[(k0 + 1) * 64 + tj * 16 + mrow];
      acc = wmma_f32k4(av, bv, acc);
    }
#pragma unroll
    for (int g = 0; g < 8; ++g) {
      const int rr = ti * 16 + hi * 8 + g, cc = tj * 16 + mrow;
      float v = acc[g];
      if (mode == 1)      v = ((rr == cc) ? beta : 0.f) - v;
      else if (mode == 2) v *= 0.25f;
      dst[rr * 64 + cc] = v;
    }
  }
  __syncthreads();
}

// ---------------- kernel 0: init global colmax scalar ---------------------
__global__ void nys_init(unsigned* colmax) { *colmax = 0u; }

// ---------------- kernel 1: landmark mean pooling -------------------------
// grid = NBH*64 blocks, 256 threads; block (bh, lm) pools 64 rows of Q and K.
__global__ __launch_bounds__(256) void nys_landmarks(const float* __restrict__ Q,
                                                     const float* __restrict__ K,
                                                     float* __restrict__ qlm,
                                                     float* __restrict__ klm) {
  __shared__ float red[256];
  const int t = threadIdx.x;
  const int bh = blockIdx.x >> 6, lm = blockIdx.x & 63;
  const size_t base = (size_t)bh * SEQ * DIM + (size_t)lm * 64 * DIM;
  const int c = t & 63, rg = t >> 6;
  float s = 0.f;
  for (int j = 0; j < 16; ++j) s += Q[base + (size_t)(rg * 16 + j) * DIM + c];
  red[t] = s;
  __syncthreads();
  if (t < 64)
    qlm[(size_t)bh * 4096 + lm * 64 + t] =
        (red[t] + red[64 + t] + red[128 + t] + red[192 + t]) * (1.f / 64.f);
  __syncthreads();
  s = 0.f;
  for (int j = 0; j < 16; ++j) s += K[base + (size_t)(rg * 16 + j) * DIM + c];
  red[t] = s;
  __syncthreads();
  if (t < 64)
    klm[(size_t)bh * 4096 + lm * 64 + t] =
        (red[t] + red[64 + t] + red[128 + t] + red[192 + t]) * (1.f / 64.f);
}

// ---------------- kernel 2: A_lm softmax + global init_coef ---------------
// Exact fp32 VALU (tiny GEMM, feeds pinv so keep full precision).
// Softmax rows sum to exactly 1 -> init_coef = global max column sum.
__global__ __launch_bounds__(256) void nys_alm(const float* __restrict__ qlm,
                                               const float* __restrict__ klm,
                                               float* __restrict__ Alm,
                                               unsigned* __restrict__ colmax) {
  __shared__ float q[4096], k[4096], a[4096];
  __shared__ float part[256], rowred[64];
  const int t = threadIdx.x, bh = blockIdx.x;
  const float4* q4 = (const float4*)(qlm + (size_t)bh * 4096);
  const float4* k4 = (const float4*)(klm + (size_t)bh * 4096);
  for (int j = 0; j < 4; ++j) {
    const int e = t + 256 * j;
    ((float4*)q)[e] = q4[e];
    ((float4*)k)[e] = k4[e];
  }
  __syncthreads();
  const int r = t >> 2, c0 = (t & 3) * 16;
  float acc[16];
#pragma unroll
  for (int i = 0; i < 16; ++i) acc[i] = 0.f;
  for (int d = 0; d < 64; ++d) {
    const float qv = q[r * 64 + d];
#pragma unroll
    for (int i = 0; i < 16; ++i) acc[i] += qv * k[(c0 + i) * 64 + d];
  }
  float lmax = -1e30f;
#pragma unroll
  for (int i = 0; i < 16; ++i) { acc[i] *= SCALE; lmax = fmaxf(lmax, acc[i]); }
  part[t] = lmax;
  __syncthreads();
  if (t < 64)
    rowred[t] = fmaxf(fmaxf(part[4 * t], part[4 * t + 1]),
                      fmaxf(part[4 * t + 2], part[4 * t + 3]));
  __syncthreads();
  const float rm = rowred[r];
  float ps = 0.f;
#pragma unroll
  for (int i = 0; i < 16; ++i) { acc[i] = __expf(acc[i] - rm); ps += acc[i]; }
  part[t] = ps;
  __syncthreads();
  if (t < 64)
    rowred[t] = part[4 * t] + part[4 * t + 1] + part[4 * t + 2] + part[4 * t + 3];
  __syncthreads();
  const float inv = 1.f / rowred[r];
#pragma unroll
  for (int i = 0; i < 16; ++i) {
    const float v = acc[i] * inv;
    a[r * 64 + c0 + i] = v;
    Alm[(size_t)bh * 4096 + r * 64 + c0 + i] = v;
  }
  __syncthreads();
  if (t < 64) {
    float cs = 0.f;
    for (int rr = 0; rr < 64; ++rr) cs += a[rr * 64 + t];
    part[t] = cs;
  } else {
    part[t] = 0.f;
  }
  __syncthreads();
  for (int off = 32; off > 0; off >>= 1) {
    if (t < off) part[t] = fmaxf(part[t], part[t + off]);
    __syncthreads();
  }
  if (t == 0) atomicMax(colmax, __float_as_uint(part[0]));  // positive floats
}

// ---------------- kernel 3a: partial BV = exp(q_lm K^T / 8) @ V -----------
// grid (BVSPLIT, NBH): each workgroup streams SEQ/64/BVSPLIT = 8 chunks.
// exp without max-subtraction (logits O(1)) => partials combine additively.
__global__ __launch_bounds__(256) void nys_bv_partial(const float* __restrict__ Kg,
                                                      const float* __restrict__ Vg,
                                                      const float* __restrict__ qlm,
                                                      float* __restrict__ pPV,
                                                      float* __restrict__ pRS) {
  __shared__ _Float16 q16[4096], kch[4096], p16[4096];
  __shared__ _Float16 vchT[64 * TP];  // V chunk transposed [d][s'], padded pitch
  __shared__ float logits[4096];
  __shared__ float rowsum[64];
  __shared__ float part[256];
  const int t = threadIdx.x, lane = t & 31, wave = t >> 5;
  const int mrow = lane & 15, hi = lane >> 4;
  const int wg = blockIdx.x, bh = blockIdx.y;
  {
    const float4* q4 = (const float4*)(qlm + (size_t)bh * 4096);
    for (int j = 0; j < 4; ++j) {
      const int e = t + 256 * j;
      const float4 v = q4[e];
      _Float16* d = q16 + e * 4;
      d[0] = (_Float16)v.x; d[1] = (_Float16)v.y;
      d[2] = (_Float16)v.z; d[3] = (_Float16)v.w;
    }
  }
  if (t < 64) rowsum[t] = 0.f;
  v8f pv[2] = {};
  __syncthreads();
  const int ch0 = wg * (64 / BVSPLIT), ch1 = ch0 + (64 / BVSPLIT);
  for (int ch = ch0; ch < ch1; ++ch) {
    const size_t base = (size_t)bh * SEQ * DIM + (size_t)ch * 64 * DIM;
    const float4* K4 = (const float4*)(Kg + base);
    const float4* V4 = (const float4*)(Vg + base);
    for (int j = 0; j < 4; ++j) {
      const int e4 = t + 256 * j;            // float4 index in 64x64 chunk
      const float4 kv = K4[e4];
      _Float16* kd = kch + e4 * 4;           // row-major [s'][d]
      kd[0] = (_Float16)kv.x; kd[1] = (_Float16)kv.y;
      kd[2] = (_Float16)kv.z; kd[3] = (_Float16)kv.w;
      const float4 vv = V4[e4];
      const int s = e4 >> 4, d0 = (e4 & 15) * 4;  // transpose into [d][s']
      vchT[(d0 + 0) * TP + s] = (_Float16)vv.x;
      vchT[(d0 + 1) * TP + s] = (_Float16)vv.y;
      vchT[(d0 + 2) * TP + s] = (_Float16)vv.z;
      vchT[(d0 + 3) * TP + s] = (_Float16)vv.w;
    }
    if (ch + 1 < ch1) {  // prefetch next chunk (global_prefetch_b8 cacheline)
      const float* nk = Kg + base + 64 * DIM;
      const float* nv = Vg + base + 64 * DIM;
      __builtin_prefetch(nk + t * 16, 0, 3);
      __builtin_prefetch(nv + t * 16, 0, 3);
    }
    __syncthreads();
    // logits[m][s'] = q_lm . K  (kch stored [s'][d] = [N][K], pitch 64)
#pragma unroll
    for (int tt = 0; tt < 2; ++tt) {
      const int idx = wave + tt * 8, ti = idx >> 2, tj = idx & 3;
      v8f acc = {};
      acc = wmma_f16(ldA(q16, ti, 0, mrow, hi), ldB_nk(kch, tj, 0, mrow, hi, 64), acc);
      acc = wmma_f16(ldA(q16, ti, 32, mrow, hi), ldB_nk(kch, tj, 32, mrow, hi, 64), acc);
#pragma unroll
      for (int g = 0; g < 8; ++g)
        logits[(ti * 16 + hi * 8 + g) * 64 + tj * 16 + mrow] = acc[g];
    }
    __syncthreads();
    const int r = t >> 2, c0 = (t & 3) * 16;
    float ps = 0.f;
#pragma unroll
    for (int i = 0; i < 16; ++i) {
      const float e = __expf(logits[r * 64 + c0 + i] * SCALE);
      p16[r * 64 + c0 + i] = (_Float16)e;
      ps += e;
    }
    part[t] = ps;
    __syncthreads();
    if (t < 64)  // deterministic fixed-order reduction (no float atomics)
      rowsum[t] += part[4 * t] + part[4 * t + 1] + part[4 * t + 2] + part[4 * t + 3];
    // PV[m][d] += P @ V  (vchT stored [d][s'] = [N][K], pitch TP)
#pragma unroll
    for (int tt = 0; tt < 2; ++tt) {
      const int idx = wave + tt * 8, ti = idx >> 2, tj = idx & 3;
      pv[tt] = wmma_f16(ldA(p16, ti, 0, mrow, hi), ldB_nk(vchT, tj, 0, mrow, hi, TP), pv[tt]);
      pv[tt] = wmma_f16(ldA(p16, ti, 32, mrow, hi), ldB_nk(vchT, tj, 32, mrow, hi, TP), pv[tt]);
    }
    __syncthreads();
  }
  const size_t obase = (size_t)(bh * BVSPLIT + wg) * 4096;
#pragma unroll
  for (int tt = 0; tt < 2; ++tt) {
    const int idx = wave + tt * 8, ti = idx >> 2, tj = idx & 3;
#pragma unroll
    for (int g = 0; g < 8; ++g) {
      const int rr = ti * 16 + hi * 8 + g, cc = tj * 16 + mrow;
      pPV[obase + rr * 64 + cc] = pv[tt][g];
    }
  }
  if (t < 64) pRS[(size_t)(bh * BVSPLIT + wg) * 64 + t] = rowsum[t];
}

// ---------------- kernel 3b: reduce partials -> BV ------------------------
__global__ __launch_bounds__(256) void nys_bv_reduce(const float* __restrict__ pPV,
                                                     const float* __restrict__ pRS,
                                                     float* __restrict__ BV) {
  __shared__ float rs[64];
  const int t = threadIdx.x, bh = blockIdx.x;
  if (t < 64) {
    float s = 0.f;
    for (int wg = 0; wg < BVSPLIT; ++wg)
      s += pRS[(size_t)(bh * BVSPLIT + wg) * 64 + t];
    rs[t] = s;
  }
  __syncthreads();
  for (int j = 0; j < 16; ++j) {
    const int e = t + 256 * j, r = e >> 6;
    float s = 0.f;
    for (int wg = 0; wg < BVSPLIT; ++wg)
      s += pPV[(size_t)(bh * BVSPLIT + wg) * 4096 + e];
    BV[(size_t)bh * 4096 + e] = s / rs[r];
  }
}

// ---------------- kernel 4: iterative pinv + C = A_tilde @ BV -------------
// All 64x64 GEMMs in LDS with exact fp32 WMMA (16x16x4).
__global__ __launch_bounds__(256) void nys_pinv(const float* __restrict__ Alm,
                                                const float* __restrict__ BV,
                                                const unsigned* __restrict__ colmax,
                                                float* __restrict__ Cm) {
  __shared__ float A[4096], Zb[4096], T1[4096], T2[4096];  // 64 KB
  const int t = threadIdx.x, bh = blockIdx.x;
  for (int j = 0; j < 4; ++j) {
    const int e = t + 256 * j;
    ((float4*)A)[e] = ((const float4*)(Alm + (size_t)bh * 4096))[e];
  }
  __syncthreads();
  const float inv = 1.f / __uint_as_float(*colmax);  // row-sum max == 1
  for (int j = 0; j < 16; ++j) {
    const int e = t + 256 * j, i = e >> 6, jj = e & 63;
    Zb[e] = A[jj * 64 + i] * inv;  // Z0 = A^T / init_coef
  }
  __syncthreads();
  float* Zp = Zb;
  float* Sp = T2;
  for (int it = 0; it < 6; ++it) {
    mm64_f32(T1, A, Zp, 1, 7.f);    // term1 = 7I  - A Z
    mm64_f32(Sp, Zp, T1, 0, 0.f);   // Z term1
    mm64_f32(T1, A, Sp, 1, 15.f);   // term2 = 15I - A (Z term1)
    mm64_f32(Sp, Zp, T1, 0, 0.f);   // Z term2
    mm64_f32(T1, A, Sp, 1, 13.f);   // term3 = 13I - A (Z term2)
    mm64_f32(Sp, Zp, T1, 2, 0.f);   // Z' = 0.25 Z term3
    float* tmp = Zp; Zp = Sp; Sp = tmp;
  }
  for (int j = 0; j < 4; ++j) {
    const int e = t + 256 * j;
    ((float4*)T1)[e] = ((const float4*)(BV + (size_t)bh * 4096))[e];
  }
  __syncthreads();
  mm64_f32(Sp, Zp, T1, 0, 0.f);     // C = A_tilde @ BV
  for (int j = 0; j < 16; ++j) {
    const int e = t + 256 * j;
    Cm[(size_t)bh * 4096 + e] = Sp[e];
  }
}

// ---------------- kernel 5: out = softmax(Q k_lm^T / 8) @ C ---------------
// grid (64 chunks, NBH); normalization folded into final epilogue.
__global__ __launch_bounds__(256) void nys_out(const float* __restrict__ Qg,
                                               const float* __restrict__ klm,
                                               const float* __restrict__ Cm,
                                               float* __restrict__ Og) {
  __shared__ _Float16 qc[4096], k16[4096], f16F[4096];
  __shared__ _Float16 c16T[64 * TP];  // C transposed [d][m], padded pitch
  __shared__ float logits[4096];
  __shared__ float part[256], rowred[64], rowsums[64];
  const int t = threadIdx.x, lane = t & 31, wave = t >> 5;
  const int mrow = lane & 15, hi = lane >> 4;
  const int ch = blockIdx.x, bh = blockIdx.y;
  const size_t qbase = (size_t)bh * SEQ * DIM + (size_t)ch * 64 * DIM;
  {
    const float4* Q4 = (const float4*)(Qg + qbase);
    const float4* L4 = (const float4*)(klm + (size_t)bh * 4096);
    const float4* C4 = (const float4*)(Cm + (size_t)bh * 4096);
    for (int j = 0; j < 4; ++j) {
      const int e4 = t + 256 * j;
      const float4 qv = Q4[e4];
      _Float16* qd = qc + e4 * 4;
      qd[0] = (_Float16)qv.x; qd[1] = (_Float16)qv.y;
      qd[2] = (_Float16)qv.z; qd[3] = (_Float16)qv.w;
      const float4 lv = L4[e4];
      _Float16* kd = k16 + e4 * 4;           // [m][d] row-major
      kd[0] = (_Float16)lv.x; kd[1] = (_Float16)lv.y;
      kd[2] = (_Float16)lv.z; kd[3] = (_Float16)lv.w;
      const float4 cv = C4[e4];
      const int m = e4 >> 4, d0 = (e4 & 15) * 4;  // transpose into [d][m]
      c16T[(d0 + 0) * TP + m] = (_Float16)cv.x;
      c16T[(d0 + 1) * TP + m] = (_Float16)cv.y;
      c16T[(d0 + 2) * TP + m] = (_Float16)cv.z;
      c16T[(d0 + 3) * TP + m] = (_Float16)cv.w;
    }
  }
  __syncthreads();
  // logits[row][m] = Q . k_lm  (k16 stored [m][d] = [N][K], pitch 64)
#pragma unroll
  for (int tt = 0; tt < 2; ++tt) {
    const int idx = wave + tt * 8, ti = idx >> 2, tj = idx & 3;
    v8f acc = {};
    acc = wmma_f16(ldA(qc, ti, 0, mrow, hi), ldB_nk(k16, tj, 0, mrow, hi, 64), acc);
    acc = wmma_f16(ldA(qc, ti, 32, mrow, hi), ldB_nk(k16, tj, 32, mrow, hi, 64), acc);
#pragma unroll
    for (int g = 0; g < 8; ++g)
      logits[(ti * 16 + hi * 8 + g) * 64 + tj * 16 + mrow] = acc[g] * SCALE;
  }
  __syncthreads();
  const int r = t >> 2, c0 = (t & 3) * 16;
  float lmax = -1e30f;
#pragma unroll
  for (int i = 0; i < 16; ++i) lmax = fmaxf(lmax, logits[r * 64 + c0 + i]);
  part[t] = lmax;
  __syncthreads();
  if (t < 64)
    rowred[t] = fmaxf(fmaxf(part[4 * t], part[4 * t + 1]),
                      fmaxf(part[4 * t + 2], part[4 * t + 3]));
  __syncthreads();
  const float rm = rowred[r];
  float ps = 0.f;
#pragma unroll
  for (int i = 0; i < 16; ++i) {
    const float e = __expf(logits[r * 64 + c0 + i] - rm);
    f16F[r * 64 + c0 + i] = (_Float16)e;  // unnormalized, in (0,1]
    ps += e;
  }
  part[t] = ps;
  __syncthreads();
  if (t < 64)
    rowsums[t] = part[4 * t] + part[4 * t + 1] + part[4 * t + 2] + part[4 * t + 3];
  __syncthreads();
  // out[row][d] = F @ C / rowsum  (c16T stored [d][m] = [N][K], pitch TP)
#pragma unroll
  for (int tt = 0; tt < 2; ++tt) {
    const int idx = wave + tt * 8, ti = idx >> 2, tj = idx & 3;
    v8f acc = {};
    acc = wmma_f16(ldA(f16F, ti, 0, mrow, hi), ldB_nk(c16T, tj, 0, mrow, hi, TP), acc);
    acc = wmma_f16(ldA(f16F, ti, 32, mrow, hi), ldB_nk(c16T, tj, 32, mrow, hi, TP), acc);
#pragma unroll
    for (int g = 0; g < 8; ++g) {
      const int rr = ti * 16 + hi * 8 + g, cc = tj * 16 + mrow;
      Og[qbase + (size_t)rr * 64 + cc] = acc[g] / rowsums[rr];
    }
  }
}

// ---------------------------------------------------------------------------
extern "C" void kernel_launch(void* const* d_in, const int* in_sizes, int n_in,
                              void* d_out, int out_size, void* d_ws, size_t ws_size,
                              hipStream_t stream) {
  (void)in_sizes; (void)n_in; (void)out_size; (void)ws_size;
  const float* Q = (const float*)d_in[0];
  const float* K = (const float*)d_in[1];
  const float* V = (const float*)d_in[2];
  float* out = (float*)d_out;

  // workspace layout (floats): qlm | klm | Alm | BV | C | pPV | pRS | colmax
  float* ws  = (float*)d_ws;
  float* qlm = ws;
  float* klm = qlm + (size_t)NBH * 4096;
  float* Alm = klm + (size_t)NBH * 4096;
  float* BV  = Alm + (size_t)NBH * 4096;
  float* Cm  = BV  + (size_t)NBH * 4096;
  float* pPV = Cm  + (size_t)NBH * 4096;
  float* pRS = pPV + (size_t)NBH * BVSPLIT * 4096;
  unsigned* colmax = (unsigned*)(pRS + (size_t)NBH * BVSPLIT * 64);

  nys_init<<<1, 1, 0, stream>>>(colmax);
  nys_landmarks<<<NBH * 64, 256, 0, stream>>>(Q, K, qlm, klm);
  nys_alm<<<NBH, 256, 0, stream>>>(qlm, klm, Alm, colmax);
  nys_bv_partial<<<dim3(BVSPLIT, NBH), 256, 0, stream>>>(K, V, qlm, pPV, pRS);
  nys_bv_reduce<<<NBH, 256, 0, stream>>>(pPV, pRS, BV);
  nys_pinv<<<NBH, 256, 0, stream>>>(Alm, BV, colmax, Cm);
  nys_out<<<dim3(64, NBH), 256, 0, stream>>>(Q, klm, Cm, out);
}
